// MultiHeadAttention_14869176778865
// MI455X (gfx1250) — matrix operand order repaired and checked by reading, not compiled, not run on hardware
//
#include <hip/hip_runtime.h>

typedef _Float16 v16h __attribute__((ext_vector_type(16)));
typedef _Float16 v8h  __attribute__((ext_vector_type(8)));
typedef float    v8f  __attribute__((ext_vector_type(8)));

#define LOG2E 1.4426950408889634f

static __device__ __forceinline__ v8f wmma_f16(v16h a, v16h b, v8f c) {
  // 8 args: (neg_a, A, neg_b, B, c_mod, C, reuse_a, reuse_b)
  return __builtin_amdgcn_wmma_f32_16x16x32_f16(false, a, false, b, (short)0, c, false, false);
}

// Build a 16-half fragment from two contiguous 8-half (16B) vector loads.
static __device__ __forceinline__ v16h frag_ld(const _Float16* p0, const _Float16* p1) {
  v8h lo = *(const v8h*)p0;
  v8h hi = *(const v8h*)p1;
  v16h r;
#pragma unroll
  for (int j = 0; j < 8; ++j) { r[j] = lo[j]; r[j + 8] = hi[j]; }
  return r;
}

// ---------------- precision conversion ----------------

__global__ void __launch_bounds__(256) cvt_f32_to_f16(const float* __restrict__ src,
                                                      _Float16* __restrict__ dst, int n) {
  int i = blockIdx.x * 256 + threadIdx.x;
  if (i < n) dst[i] = (_Float16)src[i];
}

// Convert the 4 weight matrices [1024,1024] f32 -> f16 TRANSPOSED [n][k].
// Coalesced on both sides via a padded 32x32 LDS tile.
__global__ void __launch_bounds__(256) cvt_wT(const float* __restrict__ Wq,
                                              const float* __restrict__ Wk,
                                              const float* __restrict__ Wv,
                                              const float* __restrict__ Wo,
                                              _Float16* __restrict__ WT) {
  __shared__ _Float16 tile[32][33];
  const int which = blockIdx.z;
  const float* s = (which == 0) ? Wq : (which == 1) ? Wk : (which == 2) ? Wv : Wo;
  _Float16* d = WT + (size_t)which * 1024 * 1024;

  const int kt = (blockIdx.x & 31) * 32;   // k tile origin (source rows)
  const int nt = (blockIdx.x >> 5) * 32;   // n tile origin (source cols)
  const int c = threadIdx.x & 31;
  const int r0 = threadIdx.x >> 5;         // 8 row-groups, 4 rows each

#pragma unroll
  for (int i = 0; i < 4; ++i) {
    int r = r0 * 4 + i;
    tile[r][c] = (_Float16)s[(size_t)(kt + r) * 1024 + nt + c];
  }
  __syncthreads();
#pragma unroll
  for (int i = 0; i < 4; ++i) {
    int r = r0 * 4 + i;                    // transposed row = n index
    d[(size_t)(nt + r) * 1024 + kt + c] = tile[c][r];
  }
}

// ---------------- fused QKV projection GEMM (register-blocked) ----------------
// Each wave computes a 32x64 output tile: 2 M-frags x 4 N-frags, 8 WMMAs / k-step,
// 12 b128 loads / k-step (1.5 loads per WMMA instead of 4).
// blockIdx.z selects Q/K/V; outputs scattered to head-split layouts:
//   Q,K: [B,H,S,64]   V: transposed [B,H,64,S]
__global__ void __launch_bounds__(128) gemm_qkv(const _Float16* __restrict__ xh,
                                                const _Float16* __restrict__ WT,
                                                _Float16* __restrict__ Qh,
                                                _Float16* __restrict__ Kh,
                                                _Float16* __restrict__ Vt) {
  const int lane = threadIdx.x & 31;
  const int wave = threadIdx.x >> 5;
  const int sel  = blockIdx.z;
  const int m0 = blockIdx.y * 32;
  const int n0 = (blockIdx.x * 4 + wave) * 64;

  const _Float16* W = WT + (size_t)sel * 1024 * 1024;

  const int lo16 = lane & 15;
  const int kbA  = (lane >> 4) * 8;          // A: K sub-block per half-wave
  const int kbB  = (lane >> 4) * 8;          // B: K sub-block per half-wave

  const _Float16* arow0 = xh + (size_t)(m0 + lo16) * 1024 + kbA;
  const _Float16* arow1 = arow0 + (size_t)16 * 1024;
  const _Float16* brow  = W + (size_t)(n0 + lo16) * 1024 + kbB;

  v8f acc[2][4] = {{{}, {}, {}, {}}, {{}, {}, {}, {}}};

#pragma unroll 2
  for (int k0 = 0; k0 < 1024; k0 += 32) {
    v16h a0 = frag_ld(arow0 + k0, arow0 + k0 + 16);
    v16h a1 = frag_ld(arow1 + k0, arow1 + k0 + 16);
#pragma unroll
    for (int t = 0; t < 4; ++t) {
      const _Float16* bp = brow + (size_t)t * 16 * 1024 + k0;
      v16h b = frag_ld(bp, bp + 16);
      acc[0][t] = wmma_f16(a0, b, acc[0][t]);
      acc[1][t] = wmma_f16(a1, b, acc[1][t]);
    }
  }

#pragma unroll
  for (int mi = 0; mi < 2; ++mi) {
#pragma unroll
    for (int t = 0; t < 4; ++t) {
      const int col = n0 + t * 16 + lo16;
      const int h = col >> 6, dd = col & 63;
#pragma unroll
      for (int r = 0; r < 8; ++r) {
        int m = m0 + mi * 16 + r + 8 * (lane >> 4);
        int bb = m >> 11, q = m & 2047;
        _Float16 v = (_Float16)acc[mi][t][r];
        if (sel == 0)      Qh[(((size_t)bb * 16 + h) * 2048 + q) * 64 + dd] = v;
        else if (sel == 1) Kh[(((size_t)bb * 16 + h) * 2048 + q) * 64 + dd] = v;
        else               Vt[(((size_t)bb * 16 + h) * 64 + dd) * 2048 + q] = v;
      }
    }
  }
}

// ---------------- flash attention ----------------
// One wave per (b, h, 16-query tile). Online softmax; scores never hit memory.
__global__ void __launch_bounds__(256) attn_kernel(const _Float16* __restrict__ Qh,
                                                   const _Float16* __restrict__ Kh,
                                                   const _Float16* __restrict__ Vt,
                                                   const int* __restrict__ valid_lens,
                                                   _Float16* __restrict__ ctxh) {
  __shared__ _Float16 pbuf[8 * 16 * 32];   // per-wave 16x32 f16 prob staging
  const int lane = threadIdx.x & 31;
  const int wave = threadIdx.x >> 5;
  const int g  = blockIdx.x * 8 + wave;    // [0, 4096)
  const int b  = g >> 11;                  // 2048 q-tiles per batch
  const int h  = (g >> 7) & 15;            // 128 q-tiles per head
  const int q0 = (g & 127) << 4;
  const int vl = valid_lens[b];

  const _Float16* Qb = Qh + (((size_t)b * 16 + h) * 2048) * 64;
  const _Float16* Kb = Kh + (((size_t)b * 16 + h) * 2048) * 64;
  const _Float16* Vb = Vt + (((size_t)b * 16 + h) * 64) * 2048;
  _Float16* pw = pbuf + wave * 512;

  const int lo16 = lane & 15;
  const int kbA  = (lane >> 4) * 8;
  const int hi16 = (lane >> 4) * 8;
  const int qrow = q0 + lo16;

  // Q tile 16x64 -> two A fragments (k0 = 0, 32), loaded once.
  const _Float16* qp = Qb + (size_t)qrow * 64;
  v16h qf0 = frag_ld(qp + kbA,      qp + kbA + 16);
  v16h qf1 = frag_ld(qp + 32 + kbA, qp + 32 + kbA + 16);

  v8f acc[4] = {{}, {}, {}, {}};
  float mrow[8], lrow[8];
#pragma unroll
  for (int r = 0; r < 8; ++r) { mrow[r] = -3.0e38f; lrow[r] = 0.0f; }

  for (int kt = 0; kt < 2048; kt += 32) {
    // ---- scores: two 16x16 tiles over keys [kt, kt+32) ----
    v8f s[2] = {{}, {}};
#pragma unroll
    for (int t = 0; t < 2; ++t) {
      const int key = kt + t * 16 + lo16;                 // B-operand: lane -> N (key)
      const _Float16* kp = Kb + (size_t)key * 64 + hi16;  // contiguous along d
      v16h bf0 = frag_ld(kp,      kp + 16);               // d in [0,32)
      v16h bf1 = frag_ld(kp + 32, kp + 48);               // d in [32,64)
      s[t] = wmma_f16(qf0, bf0, s[t]);
      s[t] = wmma_f16(qf1, bf1, s[t]);
      const bool valid = key < vl;
#pragma unroll
      for (int r = 0; r < 8; ++r)
        s[t][r] = valid ? s[t][r] * 0.125f : -1.0e6f;     // scale 1/sqrt(64), then mask
    }

    // ---- online softmax; rows of a C tile live inside 16-lane halves ----
#pragma unroll
    for (int r = 0; r < 8; ++r) {
      float mx = fmaxf(s[0][r], s[1][r]);
      mx = fmaxf(mx, __shfl_xor(mx, 8));
      mx = fmaxf(mx, __shfl_xor(mx, 4));
      mx = fmaxf(mx, __shfl_xor(mx, 2));
      mx = fmaxf(mx, __shfl_xor(mx, 1));
      float mn = fmaxf(mrow[r], mx);
      float alpha = exp2f((mrow[r] - mn) * LOG2E);
      mrow[r] = mn;
      float p0 = exp2f((s[0][r] - mn) * LOG2E);
      float p1 = exp2f((s[1][r] - mn) * LOG2E);
      s[0][r] = p0; s[1][r] = p1;
      float sm = p0 + p1;
      sm += __shfl_xor(sm, 8);
      sm += __shfl_xor(sm, 4);
      sm += __shfl_xor(sm, 2);
      sm += __shfl_xor(sm, 1);
      lrow[r] = lrow[r] * alpha + sm;
      acc[0][r] *= alpha; acc[1][r] *= alpha;
      acc[2][r] *= alpha; acc[3][r] *= alpha;
    }

    // ---- transpose P (C layout) -> A fragment via per-wave LDS ----
#pragma unroll
    for (int r = 0; r < 8; ++r) {
      int row = r + 8 * (lane >> 4);
      pw[row * 32 + lo16]      = (_Float16)s[0][r];
      pw[row * 32 + 16 + lo16] = (_Float16)s[1][r];
    }
    asm volatile("s_wait_dscnt 0" ::: "memory");  // DS ops are in-order per wave
    v16h pf = frag_ld(pw + lo16 * 32 + kbA, pw + lo16 * 32 + kbA + 16);

    // ---- context: acc[t] += P(16x32) x V(32x16), V pre-transposed ----
#pragma unroll
    for (int t = 0; t < 4; ++t) {
      const _Float16* vp = Vb + (size_t)(t * 16 + lo16) * 2048 + kt + hi16;
      v16h bf = frag_ld(vp, vp + 16);                     // contiguous along key
      acc[t] = wmma_f16(pf, bf, acc[t]);
    }
  }

  // ---- epilogue: normalize, write context f16 [B,S,D] head-concatenated ----
#pragma unroll
  for (int t = 0; t < 4; ++t) {
#pragma unroll
    for (int r = 0; r < 8; ++r) {
      int row = r + 8 * (lane >> 4);
      float v = acc[t][r] / lrow[r];
      ctxh[((size_t)(b * 2048 + q0 + row)) * 1024 + h * 64 + t * 16 + lo16] = (_Float16)v;
    }
  }
}

// ---------------- output projection: out = ctx @ Wo (f32 out, 32x64 tiles) ----------------
__global__ void __launch_bounds__(128) gemm_out(const _Float16* __restrict__ ctxh,
                                                const _Float16* __restrict__ WoT,
                                                float* __restrict__ out) {
  const int lane = threadIdx.x & 31;
  const int wave = threadIdx.x >> 5;
  const int m0 = blockIdx.y * 32;
  const int n0 = (blockIdx.x * 4 + wave) * 64;

  const int lo16 = lane & 15;
  const int kbA  = (lane >> 4) * 8;
  const int kbB  = (lane >> 4) * 8;

  const _Float16* arow0 = ctxh + (size_t)(m0 + lo16) * 1024 + kbA;
  const _Float16* arow1 = arow0 + (size_t)16 * 1024;
  const _Float16* brow  = WoT + (size_t)(n0 + lo16) * 1024 + kbB;

  v8f acc[2][4] = {{{}, {}, {}, {}}, {{}, {}, {}, {}}};

#pragma unroll 2
  for (int k0 = 0; k0 < 1024; k0 += 32) {
    v16h a0 = frag_ld(arow0 + k0, arow0 + k0 + 16);
    v16h a1 = frag_ld(arow1 + k0, arow1 + k0 + 16);
#pragma unroll
    for (int t = 0; t < 4; ++t) {
      const _Float16* bp = brow + (size_t)t * 16 * 1024 + k0;
      v16h b = frag_ld(bp, bp + 16);
      acc[0][t] = wmma_f16(a0, b, acc[0][t]);
      acc[1][t] = wmma_f16(a1, b, acc[1][t]);
    }
  }

#pragma unroll
  for (int mi = 0; mi < 2; ++mi) {
#pragma unroll
    for (int t = 0; t < 4; ++t) {
#pragma unroll
      for (int r = 0; r < 8; ++r) {
        int m = m0 + mi * 16 + r + 8 * (lane >> 4);
        out[(size_t)m * 1024 + n0 + t * 16 + lo16] = acc[mi][t][r];
      }
    }
  }
}

// ---------------- launch ----------------
extern "C" void kernel_launch(void* const* d_in, const int* in_sizes, int n_in,
                              void* d_out, int out_size, void* d_ws, size_t ws_size,
                              hipStream_t stream) {
  const float* x  = (const float*)d_in[0];
  const float* Wq = (const float*)d_in[1];
  const float* Wk = (const float*)d_in[2];
  const float* Wv = (const float*)d_in[3];
  const float* Wo = (const float*)d_in[4];
  const int* vlen = (const int*)d_in[5];

  // workspace layout (halves): total 24M halves = 48 MB
  _Float16* xh  = (_Float16*)d_ws;                    // 4096*1024
  _Float16* WT  = xh  + (size_t)4096 * 1024;          // 4 x 1024*1024 (transposed f16)
  _Float16* Qh  = WT  + (size_t)4 * 1024 * 1024;      // [B,H,S,64]
  _Float16* Kh  = Qh  + (size_t)4096 * 1024;          // [B,H,S,64]
  _Float16* Vt  = Kh  + (size_t)4096 * 1024;          // [B,H,64,S]
  _Float16* ctx = Vt  + (size_t)4096 * 1024;          // [B,S,D]

  cvt_f32_to_f16<<<(4096 * 1024) / 256, 256, 0, stream>>>(x, xh, 4096 * 1024);
  cvt_wT<<<dim3(32 * 32, 1, 4), 256, 0, stream>>>(Wq, Wk, Wv, Wo, WT);
  gemm_qkv<<<dim3(4, 128, 3), 128, 0, stream>>>(xh, WT, Qh, Kh, Vt);
  attn_kernel<<<512, 256, 0, stream>>>(Qh, Kh, Vt, vlen, ctx);
  gemm_out<<<dim3(4, 128), 128, 0, stream>>>(ctx, WT + (size_t)3 * 1024 * 1024,
                                             (float*)d_out);
}
